// MaskedMultiheadAttention_60730837565564
// MI455X (gfx1250) — matrix operand order repaired and checked by reading, not compiled, not run on hardware
//
#include <hip/hip_runtime.h>
#include <hip/hip_bf16.h>

// ---------------------------------------------------------------------------
// Masked multi-head attention for MI455X (gfx1250, wave32, WMMA).
// All GEMMs run on v_wmma_f32_16x16x32_bf16 with fp32 accumulation.
// ---------------------------------------------------------------------------

typedef __attribute__((ext_vector_type(16))) __bf16 bf16x16;
typedef __attribute__((ext_vector_type(8)))  __bf16 bf16x8;
typedef __attribute__((ext_vector_type(8)))  float  f32x8;

static constexpr int Dm  = 2048;   // embed dim
static constexpr int Sq  = 2048;   // sequence length
static constexpr int Hh  = 16;     // heads
static constexpr int Hd  = 128;    // head dim
static constexpr int Mt  = 4096;   // B*S tokens

static __device__ __forceinline__ __bf16 f2bf(float f) {
  // native RNE convert (v_cvt_*bf16* class on gfx1250)
  return (__bf16)f;
}

static __device__ __forceinline__ bf16x16 cat8(bf16x8 lo, bf16x8 hi) {
  return __builtin_shufflevector(lo, hi, 0,1,2,3,4,5,6,7,8,9,10,11,12,13,14,15);
}

#define WMMA_BF16(a, b, c) \
  __builtin_amdgcn_wmma_f32_16x16x32_bf16(false, (a), false, (b), (short)0, (c), false, false)

// ---------------------------------------------------------------------------
// fp32 -> bf16 weight conversion (D*D elements, 4 elems/thread, no tail)
// ---------------------------------------------------------------------------
__global__ __launch_bounds__(256) void cvt_bf16_kernel(const float* __restrict__ src,
                                                       __bf16* __restrict__ dst) {
  const size_t i = ((size_t)blockIdx.x * blockDim.x + threadIdx.x) * 4;
  float4 f = *(const float4*)(src + i);
  dst[i + 0] = f2bf(f.x);
  dst[i + 1] = f2bf(f.y);
  dst[i + 2] = f2bf(f.z);
  dst[i + 3] = f2bf(f.w);
}

// ---------------------------------------------------------------------------
// Tiled GEMM:  Out[m][n] = (sum_k X[m][k] * W[n][k] + bias[n]) * oscale
//   M=4096, N=2048, K=2048.  Block tile 128x128, BK=32, 256 threads (8 waves),
//   each wave computes 32x64 via 2x4 WMMA accumulators.
//   LDS is double-buffered: stage tile kt+1 while WMMAs consume tile kt
//   (one barrier per iteration; write/read buffers are disjoint).
// IN_BF16: 0 -> X is fp32 (converted while staging), 1 -> X is bf16.
// OUT_MODE: 0 -> bf16 [B,H,S,Hd]   (Q / K)
//           1 -> bf16 [B,H,Hd,S]   (V transposed for P*V B-fragments)
//           2 -> fp32 [M,N]        (final output)
// ---------------------------------------------------------------------------
template <int IN_BF16, int OUT_MODE>
__global__ __launch_bounds__(256, 1)
void gemm_kernel(const void* __restrict__ Xv, const __bf16* __restrict__ W,
                 const float* __restrict__ bias, void* __restrict__ Outv,
                 float oscale) {
  constexpr int LDT  = 40;        // row stride (elems): 80B, 16B-aligned, skews banks
  constexpr int TILE = 128 * LDT; // one LDS tile
  constexpr int NK   = Dm / 32;
  __shared__ __attribute__((aligned(16))) __bf16 lA[2 * TILE];
  __shared__ __attribute__((aligned(16))) __bf16 lB[2 * TILE];

  const int tid  = threadIdx.x;
  const int m0   = blockIdx.y * 128;
  const int n0   = blockIdx.x * 128;
  const int lrow = tid >> 1;            // 2 threads per staged row
  const int lcol = (tid & 1) * 16;      // 16 elems each
  const int wid  = tid >> 5;
  const int lane = tid & 31;
  const int ml   = lane & 15;
  const int half = lane >> 4;
  const int wm   = (wid & 3) * 32;      // wave row offset in block tile
  const int wn   = (wid >> 2) * 64;     // wave col offset in block tile

  auto stage = [&](int kt) {
    const int buf = kt & 1;
    __bf16* dA = &lA[buf * TILE];
    __bf16* dB = &lB[buf * TILE];
    const int k0 = kt * 32;
    // ---- stage A tile (128 x 32) ----
    if constexpr (IN_BF16) {
      const __bf16* src = (const __bf16*)Xv + (size_t)(m0 + lrow) * Dm + k0 + lcol;
      *(bf16x8*)&dA[lrow * LDT + lcol]     = *(const bf16x8*)(src);
      *(bf16x8*)&dA[lrow * LDT + lcol + 8] = *(const bf16x8*)(src + 8);
    } else {
      const float* src = (const float*)Xv + (size_t)(m0 + lrow) * Dm + k0 + lcol;
      float4 f0 = *(const float4*)(src + 0);
      float4 f1 = *(const float4*)(src + 4);
      float4 f2 = *(const float4*)(src + 8);
      float4 f3 = *(const float4*)(src + 12);
      bf16x8 a, b;
      a[0]=f2bf(f0.x); a[1]=f2bf(f0.y); a[2]=f2bf(f0.z); a[3]=f2bf(f0.w);
      a[4]=f2bf(f1.x); a[5]=f2bf(f1.y); a[6]=f2bf(f1.z); a[7]=f2bf(f1.w);
      b[0]=f2bf(f2.x); b[1]=f2bf(f2.y); b[2]=f2bf(f2.z); b[3]=f2bf(f2.w);
      b[4]=f2bf(f3.x); b[5]=f2bf(f3.y); b[6]=f2bf(f3.z); b[7]=f2bf(f3.w);
      *(bf16x8*)&dA[lrow * LDT + lcol]     = a;
      *(bf16x8*)&dA[lrow * LDT + lcol + 8] = b;
    }
    // ---- stage B tile: W rows n0..n0+127, cols k0..k0+31 (bf16) ----
    const __bf16* srcb = W + (size_t)(n0 + lrow) * Dm + k0 + lcol;
    *(bf16x8*)&dB[lrow * LDT + lcol]     = *(const bf16x8*)(srcb);
    *(bf16x8*)&dB[lrow * LDT + lcol + 8] = *(const bf16x8*)(srcb + 8);
  };

  f32x8 acc[2][4] = {};

  stage(0);
  __syncthreads();

  for (int kt = 0; kt < NK; ++kt) {
    if (kt + 1 < NK) stage(kt + 1);  // fill the other buffer during compute
    if (kt + 2 < NK) {               // prefetch hint two tiles ahead
      if constexpr (IN_BF16)
        __builtin_prefetch((const __bf16*)Xv + (size_t)(m0 + lrow) * Dm + (kt + 2) * 32 + lcol, 0, 3);
      else
        __builtin_prefetch((const float*)Xv + (size_t)(m0 + lrow) * Dm + (kt + 2) * 32 + lcol, 0, 3);
      __builtin_prefetch(W + (size_t)(n0 + lrow) * Dm + (kt + 2) * 32 + lcol, 0, 3);
    }

    const __bf16* sA = &lA[(kt & 1) * TILE];
    const __bf16* sB = &lB[(kt & 1) * TILE];

    // ---- WMMA fragments straight from LDS (contiguous k order, same for A and B) ----
    bf16x16 af[2], bfv[4];
#pragma unroll
    for (int am = 0; am < 2; ++am) {
      const __bf16* p = &sA[(wm + am * 16 + ml) * LDT];
      // A: lanes 0-15 hold K 0-15, lanes 16-31 hold K 16-31
      af[am] = cat8(*(const bf16x8*)(p + half * 16), *(const bf16x8*)(p + half * 16 + 8));
    }
#pragma unroll
    for (int bn = 0; bn < 4; ++bn) {
      const __bf16* p = &sB[(wn + bn * 16 + ml) * LDT + half * 16];
      // B: lanes 0-15 hold K 0-15, lanes 16-31 hold K 16-31 (N = lane&15)
      bfv[bn] = cat8(*(const bf16x8*)(p), *(const bf16x8*)(p + 8));
    }
#pragma unroll
    for (int am = 0; am < 2; ++am)
#pragma unroll
      for (int bn = 0; bn < 4; ++bn)
        acc[am][bn] = WMMA_BF16(af[am], bfv[bn], acc[am][bn]);

    __syncthreads();
  }

  // ---- epilogue: C layout -> lane (half,ml) holds rows half*8+r, col ml ----
#pragma unroll
  for (int am = 0; am < 2; ++am) {
#pragma unroll
    for (int bn = 0; bn < 4; ++bn) {
      const int ng = n0 + wn + bn * 16 + ml;
      const float bv = bias[ng];
#pragma unroll
      for (int r = 0; r < 8; ++r) {
        const int mg = m0 + wm + am * 16 + half * 8 + r;
        const float v = (acc[am][bn][r] + bv) * oscale;
        if constexpr (OUT_MODE == 0) {
          const int b = mg >> 11, s = mg & 2047;
          const int h = ng >> 7,  hd = ng & 127;
          ((__bf16*)Outv)[((size_t)(b * Hh + h) * Sq + s) * Hd + hd] = f2bf(v);
        } else if constexpr (OUT_MODE == 1) {
          const int b = mg >> 11, s = mg & 2047;
          const int h = ng >> 7,  hd = ng & 127;
          ((__bf16*)Outv)[((size_t)(b * Hh + h) * Hd + hd) * Sq + s] = f2bf(v);
        } else {
          ((float*)Outv)[(size_t)mg * Dm + ng] = v;
        }
      }
    }
  }
}

// ---------------------------------------------------------------------------
// Causal flash attention.  Grid: (S/64, B*H), 128 threads = 4 waves.
// Each wave owns 16 query rows; loops over 64-key tiles up to the diagonal.
// Q pre-scaled by 1/sqrt(Hd) at projection time.
// Q: [B,H,S,Hd] bf16   K: [B,H,S,Hd] bf16   Vt: [B,H,Hd,S] bf16
// Ctx out: [B,S,D] bf16 (heads concatenated)
// ---------------------------------------------------------------------------
__global__ __launch_bounds__(128, 1)
void attn_kernel(const __bf16* __restrict__ Q, const __bf16* __restrict__ K,
                 const __bf16* __restrict__ Vt, __bf16* __restrict__ Ctx) {
  __shared__ __attribute__((aligned(16))) __bf16 pT[4][16][72];  // per-wave P staging

  const int tid = threadIdx.x, wid = tid >> 5, lane = tid & 31;
  const int ml = lane & 15, half = lane >> 4;
  const int bh = blockIdx.y;                 // b*16 + h
  const int q0 = blockIdx.x * 64 + wid * 16; // this wave's first query row

  const __bf16* Qb = Q  + (size_t)bh * Sq * Hd;
  const __bf16* Kb = K  + (size_t)bh * Sq * Hd;
  const __bf16* Vb = Vt + (size_t)bh * Hd * Sq;

  // Q fragments for the whole Hd=128 (4 chunks of K=32), resident for the loop
  bf16x16 qf[4];
  {
    const __bf16* qrow = Qb + (size_t)(q0 + ml) * Hd;
#pragma unroll
    for (int c = 0; c < 4; ++c)
      qf[c] = cat8(*(const bf16x8*)(qrow + c * 32 + half * 16),
                   *(const bf16x8*)(qrow + c * 32 + half * 16 + 8));
  }

  f32x8 o[8] = {};          // 16 x 128 fp32 output accumulator (8 N-tiles)
  float mstat[8], lstat[8]; // per-row online-softmax stats (rows half*8+r)
#pragma unroll
  for (int r = 0; r < 8; ++r) { mstat[r] = -1e30f; lstat[r] = 0.0f; }

  const int nkt = (q0 >> 6) + 1;  // causal: key tiles with kbase <= q0
  for (int kt = 0; kt < nkt; ++kt) {
    const int kb = kt * 64;

    if (kt + 1 < nkt) {  // prefetch next key tile rows (hint only)
#pragma unroll
      for (int bn = 0; bn < 4; ++bn)
        __builtin_prefetch(Kb + (size_t)(kb + 64 + bn * 16 + ml) * Hd, 0, 3);
    }

    // ---- scores tile: 16 queries x 64 keys ----
    f32x8 sc[4] = {};
#pragma unroll
    for (int bn = 0; bn < 4; ++bn) {
      const __bf16* krow = Kb + (size_t)(kb + bn * 16 + ml) * Hd;
#pragma unroll
      for (int c = 0; c < 4; ++c) {
        bf16x16 kf = cat8(*(const bf16x8*)(krow + c * 32 + half * 16),
                          *(const bf16x8*)(krow + c * 32 + half * 16 + 8));
        sc[bn] = WMMA_BF16(qf[c], kf, sc[bn]);
      }
    }

    // ---- causal mask (wave-uniform branch; EXEC stays full) ----
    if (kb + 63 > q0) {
#pragma unroll
      for (int bn = 0; bn < 4; ++bn) {
        const int keyn = kb + bn * 16 + ml;
#pragma unroll
        for (int r = 0; r < 8; ++r) {
          const int qrow = q0 + half * 8 + r;
          sc[bn][r] = (keyn > qrow) ? -1e30f : sc[bn][r];
        }
      }
    }

    // ---- online softmax: row max / scale / sum (rows span a 16-lane group) ----
    float mloc[8];
#pragma unroll
    for (int r = 0; r < 8; ++r)
      mloc[r] = fmaxf(fmaxf(sc[0][r], sc[1][r]), fmaxf(sc[2][r], sc[3][r]));
#pragma unroll
    for (int off = 8; off >= 1; off >>= 1)
#pragma unroll
      for (int r = 0; r < 8; ++r)
        mloc[r] = fmaxf(mloc[r], __shfl_xor(mloc[r], off, 32));

    float scale[8];
#pragma unroll
    for (int r = 0; r < 8; ++r) {
      const float mnew = fmaxf(mstat[r], mloc[r]);
      scale[r] = __expf(mstat[r] - mnew);
      mstat[r] = mnew;
    }

    float rsum[8] = {0.f, 0.f, 0.f, 0.f, 0.f, 0.f, 0.f, 0.f};
#pragma unroll
    for (int bn = 0; bn < 4; ++bn)
#pragma unroll
      for (int r = 0; r < 8; ++r) {
        const float p = __expf(sc[bn][r] - mstat[r]);
        sc[bn][r] = p;
        rsum[r] += p;
      }
#pragma unroll
    for (int off = 8; off >= 1; off >>= 1)
#pragma unroll
      for (int r = 0; r < 8; ++r)
        rsum[r] += __shfl_xor(rsum[r], off, 32);
#pragma unroll
    for (int r = 0; r < 8; ++r) lstat[r] = lstat[r] * scale[r] + rsum[r];
#pragma unroll
    for (int j = 0; j < 8; ++j)
#pragma unroll
      for (int r = 0; r < 8; ++r) o[j][r] *= scale[r];

    // ---- P: C-layout -> A-layout via per-wave LDS (no barrier needed) ----
#pragma unroll
    for (int bn = 0; bn < 4; ++bn)
#pragma unroll
      for (int r = 0; r < 8; ++r)
        pT[wid][half * 8 + r][bn * 16 + ml] = f2bf(sc[bn][r]);

    // ---- O += P (16x64) x V (64x128) ----
#pragma unroll
    for (int c = 0; c < 2; ++c) {
      const __bf16* prow = &pT[wid][ml][0];
      bf16x16 pf = cat8(*(const bf16x8*)(prow + c * 32 + half * 16),
                        *(const bf16x8*)(prow + c * 32 + half * 16 + 8));
#pragma unroll
      for (int j = 0; j < 8; ++j) {
        const __bf16* vrow = Vb + (size_t)(j * 16 + ml) * Sq + kb + c * 32 + half * 16;
        bf16x16 vf = cat8(*(const bf16x8*)(vrow), *(const bf16x8*)(vrow + 8));
        o[j] = WMMA_BF16(pf, vf, o[j]);
      }
    }
  }

  // ---- normalize and emit context in [B,S,D] with head h at d = h*128 ----
  float inv[8];
#pragma unroll
  for (int r = 0; r < 8; ++r) inv[r] = 1.0f / fmaxf(lstat[r], 1e-20f);

  const int b = bh >> 4, h = bh & 15;
#pragma unroll
  for (int j = 0; j < 8; ++j)
#pragma unroll
    for (int r = 0; r < 8; ++r) {
      const int s = q0 + half * 8 + r;
      const int d = h * Hd + j * 16 + ml;
      Ctx[((size_t)b * Sq + s) * Dm + d] = f2bf(o[j][r] * inv[r]);
    }
}

// ---------------------------------------------------------------------------
// Launch
// ---------------------------------------------------------------------------
extern "C" void kernel_launch(void* const* d_in, const int* in_sizes, int n_in,
                              void* d_out, int out_size, void* d_ws, size_t ws_size,
                              hipStream_t stream) {
  (void)in_sizes; (void)n_in; (void)out_size; (void)ws_size;
  const float* query = (const float*)d_in[0];
  const float* key_  = (const float*)d_in[1];
  const float* value = (const float*)d_in[2];
  // d_in[3] is the tril attn_mask; causality is applied analytically.
  const float* Wq = (const float*)d_in[4];
  const float* bq = (const float*)d_in[5];
  const float* Wk = (const float*)d_in[6];
  const float* bk = (const float*)d_in[7];
  const float* Wv = (const float*)d_in[8];
  const float* bv = (const float*)d_in[9];
  const float* Wo = (const float*)d_in[10];
  const float* bo = (const float*)d_in[11];

  const size_t DD = (size_t)Dm * Dm;  // 4M elements
  char* ws = (char*)d_ws;
  __bf16* Wqb = (__bf16*)ws; ws += DD * 2;
  __bf16* Wkb = (__bf16*)ws; ws += DD * 2;
  __bf16* Wvb = (__bf16*)ws; ws += DD * 2;
  __bf16* Wob = (__bf16*)ws; ws += DD * 2;
  __bf16* Qb  = (__bf16*)ws; ws += DD * 4;  // B*S*D bf16
  __bf16* Kb  = (__bf16*)ws; ws += DD * 4;
  __bf16* Vtb = (__bf16*)ws; ws += DD * 4;
  __bf16* Ctx = (__bf16*)ws; ws += DD * 4;

  // 1) weights -> bf16
  const int cvt_blocks = (int)(DD / (256 * 4));
  cvt_bf16_kernel<<<cvt_blocks, 256, 0, stream>>>(Wq, Wqb);
  cvt_bf16_kernel<<<cvt_blocks, 256, 0, stream>>>(Wk, Wkb);
  cvt_bf16_kernel<<<cvt_blocks, 256, 0, stream>>>(Wv, Wvb);
  cvt_bf16_kernel<<<cvt_blocks, 256, 0, stream>>>(Wo, Wob);

  // 2) projections (Q pre-scaled by 1/sqrt(Hd); V stored transposed)
  dim3 gg(Dm / 128, Mt / 128);  // (16, 32)
  gemm_kernel<0, 0><<<gg, 256, 0, stream>>>(query, Wqb, bq, Qb, 0.0883883476483184f);
  gemm_kernel<0, 0><<<gg, 256, 0, stream>>>(key_,  Wkb, bk, Kb, 1.0f);
  gemm_kernel<0, 1><<<gg, 256, 0, stream>>>(value, Wvb, bv, Vtb, 1.0f);

  // 3) causal flash attention
  attn_kernel<<<dim3(Sq / 64, 2 * Hh), 128, 0, stream>>>(Qb, Kb, Vtb, Ctx);

  // 4) output projection -> fp32 d_out
  gemm_kernel<1, 2><<<gg, 256, 0, stream>>>(Ctx, Wob, bo, d_out, 1.0f);
}
